// ConvContract_23665269801343
// MI455X (gfx1250) — matrix operand order repaired and checked by reading, not compiled, not run on hardware
//
#include <hip/hip_runtime.h>

typedef __attribute__((ext_vector_type(8)))  float   v8f;
typedef __attribute__((ext_vector_type(16))) __bf16  v16bf;
typedef __attribute__((ext_vector_type(8)))  unsigned int v8u;

#define HW 128
#define CIN_TOT 288     // 96 scalar + 96*2 vector input channels
#define COUT_TOT 288    // 96 scalar + 96*2 vector output channels
#define NTAPS 9
#define NCHUNK 9        // 288 / 32 K-chunks
#define G_ELEMS (COUT_TOT * NTAPS * CIN_TOT)   // 746496
#define GSUM_ELEMS (COUT_TOT * CIN_TOT)        // 82944
#define MEANX_ELEMS (8 * CIN_TOT)              // 2304
#define MEANV_ELEMS (8 * 192)                  // 1536
#define OUT_S_ELEMS (8ull * 96 * 128 * 128)    // 12582912
// packed X: [b][r][chunk][plane][w][16 dwords]  (bf16 pairs, hi/lo planes)
#define XP_DWORDS (8ull * 128 * NCHUNK * 2 * 128 * 16)   // 37,748,736 (151 MB)
#define GP_HALVES ((size_t)COUT_TOT * NTAPS * 2 * CIN_TOT) // 1,492,992 (3 MB)

// ---- bf16 split helpers (round-to-nearest-even) ----
static __device__ __forceinline__ unsigned int f2bf(float x) {
  union { float f; unsigned int u; } v;
  v.f = x;
  unsigned int r = v.u + 0x7FFFu + ((v.u >> 16) & 1u);
  return r >> 16;
}
static __device__ __forceinline__ float bf2f(unsigned int h) {
  union { unsigned int u; float f; } v;
  v.u = h << 16;
  return v.f;
}

// ---------------------------------------------------------------------------
// K1: pack combined filter -> bf16 hi/lo halves Gp[co][tap][plane][ci]
// ---------------------------------------------------------------------------
__global__ __launch_bounds__(256) void pack_filters(
    const float* __restrict__ w_ss, const float* __restrict__ w_sv,
    const float* __restrict__ w_vs, const float* __restrict__ w_vv,
    const float* __restrict__ f_k0, const float* __restrict__ f_k1,
    const float* __restrict__ f_k2, unsigned short* __restrict__ Gp) {
  int idx = blockIdx.x * 256 + threadIdx.x;
  if (idx >= G_ELEMS) return;
  int co  = idx / (NTAPS * CIN_TOT);
  int rem = idx % (NTAPS * CIN_TOT);
  int tap = rem / CIN_TOT;
  int ci  = rem % CIN_TOT;
  float s = 0.0f;
  if (co < 96) {
    int o = co;
    if (ci < 96) {
      for (int n = 0; n < 3; ++n)
        s += w_ss[(o * 96 + ci) * 3 + n] * f_k0[n * 9 + tap];
    } else {
      int c = (ci - 96) >> 1, a = (ci - 96) & 1;
      for (int n = 0; n < 4; ++n)
        s += w_vs[(o * 96 + c) * 4 + n] * f_k1[(n * 9 + tap) * 2 + a];
    }
  } else {
    int o = (co - 96) >> 1, bb = (co - 96) & 1;
    if (ci < 96) {
      for (int n = 0; n < 4; ++n)
        s += w_sv[(o * 96 + ci) * 4 + n] * f_k1[(n * 9 + tap) * 2 + bb];
    } else {
      int c = (ci - 96) >> 1, a = (ci - 96) & 1;
      for (int n = 0; n < 5; ++n)
        s += w_vv[(o * 96 + c) * 5 + n] * f_k2[((n * 9 + tap) * 2 + a) * 2 + bb];
    }
  }
  unsigned int hi = f2bf(s);
  unsigned int lo = f2bf(s - bf2f(hi));
  size_t hbase = ((size_t)(co * NTAPS + tap) * 2) * CIN_TOT + ci;
  Gp[hbase] = (unsigned short)hi;            // plane 0 (hi)
  Gp[hbase + CIN_TOT] = (unsigned short)lo;  // plane 1 (lo)
}

// ---------------------------------------------------------------------------
// K2: Gsum[co][ci] = sum over taps (reconstructed hi+lo)
// ---------------------------------------------------------------------------
__global__ __launch_bounds__(256) void gsum_kernel(
    const unsigned short* __restrict__ Gp, float* __restrict__ Gsum) {
  int idx = blockIdx.x * 256 + threadIdx.x;
  if (idx >= GSUM_ELEMS) return;
  int co = idx / CIN_TOT, ci = idx % CIN_TOT;
  float s = 0.0f;
  for (int t = 0; t < NTAPS; ++t) {
    const unsigned short* ph = Gp + ((size_t)(co * NTAPS + t) * 2) * CIN_TOT;
    s += bf2f(ph[ci]) + bf2f(ph[CIN_TOT + ci]);
  }
  Gsum[idx] = s;
}

// ---------------------------------------------------------------------------
// K3: per-(b, folded-channel) spatial mean of the input
// ---------------------------------------------------------------------------
__global__ __launch_bounds__(256) void mean_x_kernel(
    const float* __restrict__ xs, const float* __restrict__ xv,
    float* __restrict__ meanX) {
  int blk = blockIdx.x;
  int b = blk / CIN_TOT, ci = blk % CIN_TOT;
  const float* src;
  int stride;
  if (ci < 96) {
    src = xs + (((size_t)b * 96 + ci) << 14);
    stride = 1;
  } else {
    int c = (ci - 96) >> 1, a = (ci - 96) & 1;
    src = xv + ((((size_t)b * 96 + c) << 14) << 1) + a;
    stride = 2;
  }
  float s = 0.0f;
  for (int i = threadIdx.x; i < HW * HW; i += 256)
    s += src[(size_t)i * stride];
  __shared__ float red[256];
  red[threadIdx.x] = s;
  __syncthreads();
  for (int off = 128; off > 0; off >>= 1) {
    if (threadIdx.x < off) red[threadIdx.x] += red[threadIdx.x + off];
    __syncthreads();
  }
  if (threadIdx.x == 0) meanX[blk] = red[0] * (1.0f / (HW * HW));
}

// ---------------------------------------------------------------------------
// K4: meanV[b][j] = sum_ci Gsum[96+j][ci] * meanX[b][ci]
// ---------------------------------------------------------------------------
__global__ __launch_bounds__(256) void meanv_kernel(
    const float* __restrict__ Gsum, const float* __restrict__ meanX,
    float* __restrict__ meanV) {
  int j = blockIdx.x * 256 + threadIdx.x;
  if (j >= MEANV_ELEMS) return;
  int b = j / 192, t = j % 192;
  const float* gs = Gsum + (size_t)(96 + t) * CIN_TOT;
  const float* mx = meanX + b * CIN_TOT;
  float s = 0.0f;
  for (int ci = 0; ci < CIN_TOT; ++ci) s += gs[ci] * mx[ci];
  meanV[j] = s;
}

// ---------------------------------------------------------------------------
// K5: pre-split X into bf16 hi/lo planes in LDS-tile order:
// Xp[b][r][chunk][plane][w][16 dwords], dword j of row w = halves (2j,2j+1)
// of the chunk's 32 channels.  One block per (b,r); LDS transpose per chunk
// so global writes are fully-coalesced b128 stores.
// ---------------------------------------------------------------------------
__global__ __launch_bounds__(256) void xpack_kernel(
    const float* __restrict__ xs, const float* __restrict__ xv,
    unsigned int* __restrict__ Xp) {
  __shared__ unsigned int LT[2 * 128 * 16];   // [plane][w][16 dwords] = 16 KB
  const int blk = blockIdx.x;                 // 0..1023
  const int b = blk >> 7, r = blk & 127;
  const int t = threadIdx.x;
  unsigned short* lth = (unsigned short*)LT;

  for (int chunk = 0; chunk < NCHUNK; ++chunk) {
    __syncthreads();
#pragma unroll
    for (int k = 0; k < 16; ++k) {
      int e = t + (k << 8);                  // 4096 = 32 ci x 128 w
      int ciL = e >> 7, w = e & 127;
      int cc = chunk * 32 + ciL;
      float f;
      if (cc < 96) {
        f = xs[((((size_t)b * 96 + cc) << 7) + r) * HW + w];
      } else {
        int c = (cc - 96) >> 1, a = (cc - 96) & 1;
        f = xv[((((((size_t)b * 96 + c) << 7) + r) * HW) << 1) + (w << 1) + a];
      }
      unsigned int hi = f2bf(f);
      unsigned int lo = f2bf(f - bf2f(hi));
      lth[(w << 5) + ciL] = (unsigned short)hi;            // plane 0
      lth[(128 << 5) + (w << 5) + ciL] = (unsigned short)lo;  // plane 1
    }
    __syncthreads();
    size_t base = (((size_t)b * 128 + r) * NCHUNK + chunk) * (2 * 128 * 16);
#pragma unroll
    for (int q = 0; q < 4; ++q) {
      int o = (t << 2) + (q << 10);
      *(uint4*)(Xp + base + o) = *(const uint4*)(LT + o);
    }
  }
}

// ---------------------------------------------------------------------------
// K6: implicit-GEMM torus conv, bf16x3 split on v_wmma_f32_16x16x32_bf16.
// Grid: x = B*H rows (1024), y = 9 N-tiles of 32.  Block 256 = 8 waves
// (4M x 2N); wave = 32M x 16N.  Double-buffered LDS, software-pipelined
// pure-copy staging (no conversions, no divergence in the hot loop).
// ---------------------------------------------------------------------------
#define AROW 20          // dwords per A row (16 data + 4 pad, 80 B stride)
#define BROW 20
#define A_PLANE (128 * AROW)       // 2560 dwords
#define B_OFF (2 * A_PLANE)        // 5120
#define B_PLANE (32 * BROW)        // 640
#define BUF_DW (B_OFF + 2 * B_PLANE)   // 6400 dwords / buffer (25.6 KB)
#define NSTAGE (NTAPS * NCHUNK)    // 81

__global__ __launch_bounds__(256) void conv_main(
    const unsigned int* __restrict__ Xp, const unsigned int* __restrict__ Gp32,
    const float* __restrict__ meanV, const float* __restrict__ b_s,
    const float* __restrict__ b_v, float* __restrict__ out) {
  __shared__ __align__(16) unsigned int SH[2][BUF_DW];

  const int row = blockIdx.x;            // 0..1023
  const int b = row >> 7, h = row & 127;
  const int co0 = blockIdx.y * 32;
  const int t = threadIdx.x;
  const int lane = t & 31;
  const int wv = t >> 5;
  const int waveM = wv & 3, waveN = wv >> 2;
  const int mb = waveM * 32;
  const int nn = lane & 15;
  const int kh = lane >> 4;

  // copy roles: A = (plane, w) with 4 b128 each; B = (n, plane, quad)
  const int ap = t >> 7, aw = t & 127;
  const int bn = t >> 3, bp = (t >> 2) & 1, bq = t & 3;

  v8f acc0 = {};
  v8f acc1 = {};

  uint4 ra0, ra1, ra2, ra3, rb;
  auto load_stage = [&](int s) {
    int tap = s / NCHUNK, chunk = s - tap * NCHUNK;
    int r = (h + tap / 3 - 1 + HW) & (HW - 1);
    const unsigned int* asrc =
        Xp + (((((size_t)b * 128 + r) * NCHUNK + chunk) * 2 + ap) * 128 + aw) * 16;
    ra0 = *(const uint4*)(asrc + 0);
    ra1 = *(const uint4*)(asrc + 4);
    ra2 = *(const uint4*)(asrc + 8);
    ra3 = *(const uint4*)(asrc + 12);
    const unsigned int* bsrc =
        Gp32 + ((size_t)((co0 + bn) * NTAPS + tap) * 2 + bp) * (CIN_TOT / 2) +
        chunk * 16 + bq * 4;
    rb = *(const uint4*)bsrc;
  };
  auto store_stage = [&](int buf) {
    unsigned int* d = &SH[buf][0];
    unsigned int* arow = d + ap * A_PLANE + aw * AROW;
    *(uint4*)(arow + 0) = ra0;
    *(uint4*)(arow + 4) = ra1;
    *(uint4*)(arow + 8) = ra2;
    *(uint4*)(arow + 12) = ra3;
    *(uint4*)(d + B_OFF + bp * B_PLANE + bn * BROW + bq * 4) = rb;
  };

  load_stage(0);
  store_stage(0);
  int sel = 0;

  for (int s = 0; s < NSTAGE; ++s) {
    const int tap = s / NCHUNK;
    const int dx = tap % 3 - 1;
    // kick off next stage's global loads (latency hidden behind WMMA)
    load_stage(s < NSTAGE - 1 ? s + 1 : NSTAGE - 1);
    __syncthreads();                      // buf[sel] fully staged

    const unsigned int* AsHi = &SH[sel][0];
    const unsigned int* AsLo = AsHi + A_PLANE;
    const unsigned int* BsHi = &SH[sel][B_OFF];
    const unsigned int* BsLo = BsHi + B_PLANE;

    const int wa0 = (mb + nn + dx + HW) & (HW - 1);
    const int wa1 = (mb + 16 + nn + dx + HW) & (HW - 1);
    const int aoff = kh * 4;
    const int boff = kh * 4;

    v8u u;
    uint4 q0, q1;
    q0 = *(const uint4*)&AsHi[wa0 * AROW + aoff];
    q1 = *(const uint4*)&AsHi[wa0 * AROW + 8 + aoff];
    u = (v8u){q0.x, q0.y, q0.z, q0.w, q1.x, q1.y, q1.z, q1.w};
    v16bf aH0 = __builtin_bit_cast(v16bf, u);
    q0 = *(const uint4*)&AsHi[wa1 * AROW + aoff];
    q1 = *(const uint4*)&AsHi[wa1 * AROW + 8 + aoff];
    u = (v8u){q0.x, q0.y, q0.z, q0.w, q1.x, q1.y, q1.z, q1.w};
    v16bf aH1 = __builtin_bit_cast(v16bf, u);
    q0 = *(const uint4*)&AsLo[wa0 * AROW + aoff];
    q1 = *(const uint4*)&AsLo[wa0 * AROW + 8 + aoff];
    u = (v8u){q0.x, q0.y, q0.z, q0.w, q1.x, q1.y, q1.z, q1.w};
    v16bf aL0 = __builtin_bit_cast(v16bf, u);
    q0 = *(const uint4*)&AsLo[wa1 * AROW + aoff];
    q1 = *(const uint4*)&AsLo[wa1 * AROW + 8 + aoff];
    u = (v8u){q0.x, q0.y, q0.z, q0.w, q1.x, q1.y, q1.z, q1.w};
    v16bf aL1 = __builtin_bit_cast(v16bf, u);
    const int nrow = (waveN * 16 + nn) * BROW;
    q0 = *(const uint4*)&BsHi[nrow + boff];
    q1 = *(const uint4*)&BsHi[nrow + boff + 8];
    u = (v8u){q0.x, q0.y, q0.z, q0.w, q1.x, q1.y, q1.z, q1.w};
    v16bf bH = __builtin_bit_cast(v16bf, u);
    q0 = *(const uint4*)&BsLo[nrow + boff];
    q1 = *(const uint4*)&BsLo[nrow + boff + 8];
    u = (v8u){q0.x, q0.y, q0.z, q0.w, q1.x, q1.y, q1.z, q1.w};
    v16bf bL = __builtin_bit_cast(v16bf, u);

    acc0 = __builtin_amdgcn_wmma_f32_16x16x32_bf16(
        false, aH0, false, bH, (short)0, acc0, false, false);
    acc1 = __builtin_amdgcn_wmma_f32_16x16x32_bf16(
        false, aH1, false, bH, (short)0, acc1, false, false);
    acc0 = __builtin_amdgcn_wmma_f32_16x16x32_bf16(
        false, aH0, false, bL, (short)0, acc0, false, false);
    acc1 = __builtin_amdgcn_wmma_f32_16x16x32_bf16(
        false, aH1, false, bL, (short)0, acc1, false, false);
    acc0 = __builtin_amdgcn_wmma_f32_16x16x32_bf16(
        false, aL0, false, bH, (short)0, acc0, false, false);
    acc1 = __builtin_amdgcn_wmma_f32_16x16x32_bf16(
        false, aL1, false, bH, (short)0, acc1, false, false);

    store_stage(sel ^ 1);                 // readers of buf[sel^1] are all
    sel ^= 1;                             // past the barrier above
  }

  // ---- epilogue: bias (scalar) or mean-scaled bias (vector) + store ----
  const int co = co0 + waveN * 16 + nn;
  const int mhalf = (lane >> 4) * 8;
  float addv;
  size_t base;
  int stride;
  if (co < 96) {
    addv = b_s[co];
    base = ((((size_t)b * 96 + co) << 7) + h) * HW;
    stride = 1;
  } else {
    int o = (co - 96) >> 1;
    addv = b_v[o] * meanV[b * 192 + (co - 96)];
    base = OUT_S_ELEMS + (((((size_t)b * 96 + o) << 7) + h) * HW) * 2 +
           ((co - 96) & 1);
    stride = 2;
  }
#pragma unroll
  for (int rr = 0; rr < 8; ++rr) {
    int m0 = mb + rr + mhalf;
    int m1 = mb + 16 + rr + mhalf;
    out[base + (size_t)m0 * stride] = acc0[rr] + addv;
    out[base + (size_t)m1 * stride] = acc1[rr] + addv;
  }
}

// ---------------------------------------------------------------------------
extern "C" void kernel_launch(void* const* d_in, const int* in_sizes, int n_in,
                              void* d_out, int out_size, void* d_ws,
                              size_t ws_size, hipStream_t stream) {
  const float* xs  = (const float*)d_in[0];
  const float* xv  = (const float*)d_in[1];
  const float* fk0 = (const float*)d_in[2];
  const float* fk1 = (const float*)d_in[3];
  const float* fk2 = (const float*)d_in[4];
  const float* wss = (const float*)d_in[5];
  const float* wsv = (const float*)d_in[6];
  const float* wvs = (const float*)d_in[7];
  const float* wvv = (const float*)d_in[8];
  const float* bs  = (const float*)d_in[9];
  const float* bv  = (const float*)d_in[10];
  float* out = (float*)d_out;

  // workspace: Xp (151 MB) | Gp (3 MB) | Gsum | meanX | meanV  (~155 MB)
  unsigned int*   Xp    = (unsigned int*)d_ws;
  unsigned short* Gp    = (unsigned short*)(Xp + XP_DWORDS);
  float*          Gsum  = (float*)(Gp + GP_HALVES);
  float*          meanX = Gsum + GSUM_ELEMS;
  float*          meanV = meanX + MEANX_ELEMS;

  pack_filters<<<G_ELEMS / 256, 256, 0, stream>>>(wss, wsv, wvs, wvv, fk0, fk1,
                                                  fk2, Gp);
  gsum_kernel<<<GSUM_ELEMS / 256, 256, 0, stream>>>(Gp, Gsum);
  mean_x_kernel<<<MEANX_ELEMS, 256, 0, stream>>>(xs, xv, meanX);
  meanv_kernel<<<(MEANV_ELEMS + 255) / 256, 256, 0, stream>>>(Gsum, meanX,
                                                              meanV);
  xpack_kernel<<<8 * HW, 256, 0, stream>>>(xs, xv, Xp);
  dim3 grid(8 * HW, COUT_TOT / 32);     // 1024 image rows x 9 N-tiles
  conv_main<<<grid, 256, 0, stream>>>(Xp, (const unsigned int*)Gp, meanV, bs,
                                      bv, out);
}
